// EgoRNN_89421219103090
// MI455X (gfx1250) — compile-verified, matrix operand order checked by reading, not run on hardware
//
#include <hip/hip_runtime.h>
#include <hip/hip_bf16.h>
#include <math.h>

// EgoRNN for MI455X (gfx1250, wave32, WMMA), fragment-resident version.
//
// All GEMM operands live in global memory PRE-SWIZZLED into the ISA §7.12.2
// per-lane WMMA fragment layout, so the inner loops are pure
//   global_load_b128 x4  +  v_wmma_f32_16x16x32_f16
// with no LDS staging, no barriers, no per-element shuffles.
//
// Fragment layout (16-bit, wave32): one 16x32 A-tile (or 32x16 B-tile) is
// 32 lanes x 16 halves, stored contiguously:  frag[f][lane][h], 1KB/fragment.
//   A (m,k):  lane = (m&15) + (((k>>3)&1)<<4) ; h = ((k>>4)<<3) + (k&7)
//   B (k,n):  lane = (n&15) + (((k>>3)&1)<<4) ; h = ((k>>4)<<3) + (k&7)
//   C/D f32:  m = v + ((lane>=16)?8:0), n = lane&15

#define BS 64
#define TS 256
#define NODES 1024
#define ONODES 512
#define BETA 10.0f

typedef __attribute__((ext_vector_type(16))) _Float16 v16h;
typedef __attribute__((ext_vector_type(8)))  float    v8f;

__device__ __forceinline__ int frag_lane_a(int mr, int kr) {
    return (mr & 15) + (((kr >> 3) & 1) << 4);
}
__device__ __forceinline__ int frag_h(int kr) {
    return ((kr >> 4) << 3) + (kr & 7);
}

// ---------- one-time weight swizzles ----------
// U (1024x1024, k-major) -> B-fragments: Uf[nt*32 + kt][lane][h], nt 0..63
__global__ void u_to_frag_kernel(const float* __restrict__ U, _Float16* __restrict__ Uf) {
    int gid = blockIdx.x * 256 + threadIdx.x;       // 1024*1024 threads
    int k = gid >> 10, n = gid & 1023;
    int kt = k >> 5, kr = k & 31, nt = n >> 4;
    int l = frag_lane_a(n, kr);
    int h = frag_h(kr);
    Uf[(size_t)(((nt * 32 + kt) * 32) + l) * 16 + h] = (_Float16)U[gid];
}

// W_out (1024x512) -> B-fragments: Wf[nt*32 + kt][lane][h], nt 0..31
__global__ void w_to_frag_kernel(const float* __restrict__ W, _Float16* __restrict__ Wf) {
    int gid = blockIdx.x * 256 + threadIdx.x;       // 1024*512 threads
    int k = gid >> 9, n = gid & 511;
    int kt = k >> 5, kr = k & 31, nt = n >> 4;
    int l = frag_lane_a(n, kr);
    int h = frag_h(kr);
    Wf[(size_t)(((nt * 32 + kt) * 32) + l) * 16 + h] = (_Float16)W[gid];
}

// start_state (1x1024) broadcast to H0 (64x1024) -> A-fragments
__global__ void h0_to_frag_kernel(const float* __restrict__ s0, _Float16* __restrict__ Hf) {
    int gid = blockIdx.x * 256 + threadIdx.x;       // 64*1024 threads
    int m = gid >> 10, k = gid & 1023;
    int mt = m >> 4, mr = m & 15, kt = k >> 5, kr = k & 31;
    int l = frag_lane_a(mr, kr);
    int h = frag_h(kr);
    Hf[(size_t)(((mt * 32 + kt) * 32) + l) * 16 + h] = (_Float16)s0[k];
}

// ---------- input projection (K=2, VALU) : xw[t][b][n] ----------
__global__ void input_proj_kernel(const float* __restrict__ x,
                                  const float* __restrict__ W_in,
                                  const float* __restrict__ bias,
                                  float* __restrict__ xw) {
    int gid = blockIdx.x * 256 + threadIdx.x;       // 256*64*1024 threads
    int n  = gid & 1023;
    int bt = gid >> 10;          // t*64 + b
    int b  = bt & 63;
    int t  = bt >> 6;
    float x0 = x[(b * TS + t) * 2 + 0];
    float x1 = x[(b * TS + t) * 2 + 1];
    xw[gid] = x0 * W_in[n] + x1 * W_in[NODES + n] + bias[n];
}

// ---------- recurrent step: H_t = tanh(XW_t + H_{t-1} @ U) ----------
// grid = 32 blocks, block = 256 (8 waves: 4 m-tiles x 2 n-tiles).
// Pure fragment loads; epilogue writes next-step A-frags + Sfrag for phase 3.
__global__ void rnn_step_kernel(const _Float16* __restrict__ Hf_in,
                                const _Float16* __restrict__ Uf,
                                const float* __restrict__ xw_t,
                                _Float16* __restrict__ Hf_out,
                                _Float16* __restrict__ Sfrag,
                                int t) {
    int lane = threadIdx.x & 31;
    int wave = threadIdx.x >> 5;
    int mt = wave & 3;                       // 0..3  (M = 64)
    int nt = blockIdx.x * 2 + (wave >> 2);   // 0..63 (N = 1024)

    const v16h* Ab = ((const v16h*)Hf_in) + (size_t)(mt * 32) * 32 + lane;
    const v16h* Bb = ((const v16h*)Uf)    + (size_t)(nt * 32) * 32 + lane;

    v8f acc = {};
#pragma unroll 8
    for (int kt = 0; kt < 32; ++kt) {
        v16h a  = Ab[kt * 32];
        v16h bf = Bb[kt * 32];
        acc = __builtin_amdgcn_wmma_f32_16x16x32_f16(
            false, a, false, bf, (short)0, acc, false, false);
    }

    // epilogue: tanh(acc + xw), scatter to fragment-ordered buffers
    int nc  = nt * 16 + (lane & 15);         // node column 0..1023
    int kt2 = nc >> 5, kr = nc & 31;
    int h2  = frag_h(kr);
    int hp  = ((kr >> 3) & 1) << 4;          // lane-group select from kr
    int Mtt = t >> 4, Mr = t & 15;           // phase-3 row pieces
    int l3  = Mr + hp;
#pragma unroll
    for (int v = 0; v < 8; ++v) {
        int m = mt * 16 + ((lane >> 4) << 3) + v;     // batch row 0..63
        float val = tanhf(acc[v] + xw_t[m * NODES + nc]);
        _Float16 hv = (_Float16)val;
        // next-step A-frag: tile (m>>4, kt2), lane from (m&15, kr)
        int l2 = (m & 15) + hp;
        Hf_out[(size_t)((((m >> 4) * 32 + kt2) * 32) + l2) * 16 + h2] = hv;
        // phase-3 A-frag: global row M = m*256 + t -> tile Mt = m*16 + (t>>4)
        int Mt = m * 16 + Mtt;
        Sfrag[(size_t)(((size_t)(Mt * 32 + kt2) * 32) + l3) * 16 + h2] = hv;
    }
}

// ---------- dense output GEMM: logits = relu(S @ W_out + b_out) * BETA ----
// M = 16384, N = 512, K = 1024.  grid = (256, 16), block = 256
// (8 waves: 4 M-tiles x 2 N-tiles). Pure fragment loads, no LDS.
__global__ void dense_out_kernel(const _Float16* __restrict__ Sfrag,
                                 const _Float16* __restrict__ Wf,
                                 const float* __restrict__ b_out,
                                 float* __restrict__ logits) {
    int lane = threadIdx.x & 31;
    int wave = threadIdx.x >> 5;
    int Mt = blockIdx.x * 4 + (wave & 3);    // 0..1023
    int nt = blockIdx.y * 2 + (wave >> 2);   // 0..31

    const v16h* Ab = ((const v16h*)Sfrag) + (size_t)Mt * 32 * 32 + lane;
    const v16h* Bb = ((const v16h*)Wf)    + (size_t)(nt * 32) * 32 + lane;

    v8f acc = {};
#pragma unroll 8
    for (int kt = 0; kt < 32; ++kt) {
        v16h a  = Ab[kt * 32];
        v16h bf = Bb[kt * 32];
        acc = __builtin_amdgcn_wmma_f32_16x16x32_f16(
            false, a, false, bf, (short)0, acc, false, false);
    }

    int nc = nt * 16 + (lane & 15);
    float bo = b_out[nc];
#pragma unroll
    for (int v = 0; v < 8; ++v) {
        int mr = Mt * 16 + ((lane >> 4) << 3) + v;    // global row b*256+t
        float val = acc[v] + bo;
        val = val > 0.0f ? val : 0.0f;                // relu
        logits[(size_t)mr * ONODES + nc] = val * BETA;
    }
}

// ---------- softmax over time + expected centers ----------
__global__ void ta_centers_kernel(const float* __restrict__ logits,
                                  const float* __restrict__ r,
                                  float* __restrict__ centers) {
    int gid = blockIdx.x * 256 + threadIdx.x;     // 64*512 threads
    if (gid >= BS * ONODES) return;
    int b = gid >> 9, n = gid & 511;
    const float* L = logits + (size_t)b * TS * ONODES + n;

    float mx = -3.4e38f;
    for (int t = 0; t < TS; ++t) mx = fmaxf(mx, L[t * ONODES]);
    float s = 0.f, c0 = 0.f, c1 = 0.f;
    for (int t = 0; t < TS; ++t) {
        float e = __expf(L[t * ONODES] - mx);
        s  += e;
        c0 += e * r[(b * TS + t) * 2 + 0];
        c1 += e * r[(b * TS + t) * 2 + 1];
    }
    centers[gid * 2 + 0] = c0 / s;
    centers[gid * 2 + 1] = c1 / s;
}

// ---------- softmax over units + expected position ----------
__global__ void pa_position_kernel(const float* __restrict__ logits,
                                   const float* __restrict__ centers,
                                   float* __restrict__ out) {
    int wid  = blockIdx.x * 8 + (threadIdx.x >> 5);   // b*256 + t
    int lane = threadIdx.x & 31;
    const float* L = logits + (size_t)wid * ONODES;
    int b = wid >> 8;

    float mx = -3.4e38f;
#pragma unroll
    for (int i = 0; i < 16; ++i) mx = fmaxf(mx, L[lane + i * 32]);
    for (int off = 16; off > 0; off >>= 1)
        mx = fmaxf(mx, __shfl_xor(mx, off, 32));

    float s = 0.f, p0 = 0.f, p1 = 0.f;
#pragma unroll
    for (int i = 0; i < 16; ++i) {
        int n = lane + i * 32;
        float e = __expf(L[n] - mx);
        s  += e;
        p0 += e * centers[(b * ONODES + n) * 2 + 0];
        p1 += e * centers[(b * ONODES + n) * 2 + 1];
    }
    for (int off = 16; off > 0; off >>= 1) {
        s  += __shfl_xor(s,  off, 32);
        p0 += __shfl_xor(p0, off, 32);
        p1 += __shfl_xor(p1, off, 32);
    }
    if (lane == 0) {
        out[wid * 2 + 0] = p0 / s;
        out[wid * 2 + 1] = p1 / s;
    }
}

extern "C" void kernel_launch(void* const* d_in, const int* in_sizes, int n_in,
                              void* d_out, int out_size, void* d_ws, size_t ws_size,
                              hipStream_t stream) {
    const float* x           = (const float*)d_in[0];  // (64,256,2)
    const float* r           = (const float*)d_in[1];  // (64,256,2)
    const float* W_in        = (const float*)d_in[2];  // (2,1024)
    const float* U           = (const float*)d_in[3];  // (1024,1024)
    const float* bias        = (const float*)d_in[4];  // (1024,)
    const float* start_state = (const float*)d_in[5];  // (1,1024)
    const float* W_out       = (const float*)d_in[6];  // (1024,512)
    const float* b_out       = (const float*)d_in[7];  // (512,)
    float* out = (float*)d_out;                        // (64,256,2)

    // workspace layout (bytes)
    char* ws = (char*)d_ws;
    float*    xw      = (float*)ws;                          // 64 MB  [t][b][n]
    _Float16* Uf      = (_Float16*)(ws + 67108864);          //  2 MB  B-frags
    _Float16* Wf      = (_Float16*)(ws + 69206016);          //  1 MB  B-frags
    _Float16* Hf0     = (_Float16*)(ws + 70254592);          // 128 KB A-frags (ping)
    _Float16* Hf1     = (_Float16*)(ws + 70385664);          // 128 KB A-frags (pong)
    _Float16* Sfrag   = (_Float16*)(ws + 70516736);          // 32 MB  A-frags (all t)
    float*    logits  = (float*)(ws + 104071168);            // 32 MB
    float*    centers = (float*)(ws + 137625600);            // 256 KB

    // one-time swizzles + input projection
    u_to_frag_kernel<<<(NODES * NODES) / 256, 256, 0, stream>>>(U, Uf);
    w_to_frag_kernel<<<(NODES * ONODES) / 256, 256, 0, stream>>>(W_out, Wf);
    h0_to_frag_kernel<<<(BS * NODES) / 256, 256, 0, stream>>>(start_state, Hf0);
    input_proj_kernel<<<(TS * BS * NODES) / 256, 256, 0, stream>>>(x, W_in, bias, xw);

    // sequential recurrent scan: 256 small WMMA GEMMs, ping-pong H fragments
    _Float16* hbuf[2] = { Hf0, Hf1 };
    for (int t = 0; t < TS; ++t) {
        rnn_step_kernel<<<NODES / 32, 256, 0, stream>>>(
            hbuf[t & 1], Uf, xw + (size_t)t * BS * NODES, hbuf[(t + 1) & 1], Sfrag, t);
    }

    // dense output GEMM -> logits
    dense_out_kernel<<<dim3((BS * TS) / 64, ONODES / 32), 256, 0, stream>>>(
        Sfrag, Wf, b_out, logits);

    // softmax over time + centers
    ta_centers_kernel<<<(BS * ONODES + 255) / 256, 256, 0, stream>>>(logits, r, centers);

    // softmax over units + expected position
    pa_position_kernel<<<(BS * TS) / 8, 256, 0, stream>>>(logits, centers, out);
}
